// DGALayer_24464133718852
// MI455X (gfx1250) — matrix operand order repaired and checked by reading, not compiled, not run on hardware
//
#include <hip/hip_runtime.h>
#include <hip/hip_bf16.h>

// ---------------------------------------------------------------------------
// DGA layer for MI455X (gfx1250): bf16 WMMA GEMMs + LDS bitonic top-k +
// fused bilinear box-attention sampling.  wave32 throughout.
// All GEMMs use a single contiguous-operand WMMA kernel; the attention V
// operand is pre-transposed through LDS so no strided-gather GEMM path exists.
// ---------------------------------------------------------------------------

typedef __attribute__((ext_vector_type(16))) __bf16 v16bf;
typedef __attribute__((ext_vector_type(8)))  float  v8f;

constexpr int CB   = 2;
constexpr int CH   = 128;
constexpr int CW   = 128;
constexpr int CHW  = CH * CW;      // 16384
constexpr int CD   = 256;
constexpr int CNH  = 8;
constexpr int CHD  = CD / CNH;     // 32
constexpr int CNP  = 25;
constexpr int CFG  = 4096;         // ceil(16384 * 0.25)
constexpr int CQN  = 1024;         // ceil(4096 * 0.25)
constexpr int CDFF = 1024;

// float -> bf16 via the native conversion path (lowered to packed
// v_cvt_pk_bf16_f32 on gfx1250 -- 2 elements/instruction, branchless).
__device__ __forceinline__ __bf16 f2bf(float x) { return (__bf16)x; }

// ---------------------------------------------------------------------------
// Batched WMMA GEMM:  C[z] = A[z](MxK) * B[z]^T + bias, optional ReLU.
// B is weight-style [N,K] row-major (ldb between N rows).
// z-offset = (z/nz2)*s?1 + (z%nz2)*s?2  (grid.z covers (batch, head)).
// One wave computes a 16x32 output tile: two v_wmma_f32_16x16x32_bf16 share a
// single A fragment (halves A loads/conversions per WMMA, two independent
// accumulator chains pipeline across WMMA latency).  8 waves tile 128 rows.
// Requirements: K % 32 == 0, M % 16 == 0 (true for every GEMM here).
// ---------------------------------------------------------------------------
__global__ __launch_bounds__(256)
void wmma_gemm_kernel(const float* __restrict__ A, int lda,
                      const float* __restrict__ Bm, int ldb,
                      const float* __restrict__ bias,
                      float* __restrict__ C, int ldc,
                      int M, int N, int K, int relu,
                      int nz2,
                      long long sA1, long long sA2,
                      long long sB1, long long sB2,
                      long long sC1, long long sC2)
{
  const int z = blockIdx.z;
  A  += (long long)(z / nz2) * sA1 + (long long)(z % nz2) * sA2;
  Bm += (long long)(z / nz2) * sB1 + (long long)(z % nz2) * sB2;
  C  += (long long)(z / nz2) * sC1 + (long long)(z % nz2) * sC2;

  const int lane = threadIdx.x & 31;
  const int wave = threadIdx.x >> 5;
  const int row0 = (blockIdx.x * 8 + wave) * 16;
  const int col0 = blockIdx.y * 32;
  if (row0 >= M) return;                      // whole-wave uniform exit

  const int half = lane >> 4;                 // 0 or 1
  const int l15  = lane & 15;
  const int aRow = row0 + l15;                // A frag: M in lanes (always valid)
  const int aKb  = half * 16;                 // K sub-block base within 32
  const int bKb  = half * 16;
  const int bCol0 = col0 + l15;               // B frags: N in lanes
  const int bCol1 = col0 + 16 + l15;
  const bool bValid0 = (bCol0 < N);
  const bool bValid1 = (bCol1 < N);
  const __bf16 zb = (__bf16)0.0f;

  const float* aBase = A + (long long)aRow * lda + aKb;

  v8f acc0 = {}, acc1 = {};
  for (int k0 = 0; k0 < K; k0 += 32) {
    v16bf af, bf0, bf1;
    {
      const float* ap = aBase + k0;
      float4 a0 = *(const float4*)(ap);
      float4 a1 = *(const float4*)(ap + 4);
      float4 a2 = *(const float4*)(ap + 8);
      float4 a3 = *(const float4*)(ap + 12);
      af[0] = f2bf(a0.x); af[1] = f2bf(a0.y); af[2]  = f2bf(a0.z); af[3]  = f2bf(a0.w);
      af[4] = f2bf(a1.x); af[5] = f2bf(a1.y); af[6]  = f2bf(a1.z); af[7]  = f2bf(a1.w);
      af[8] = f2bf(a2.x); af[9] = f2bf(a2.y); af[10] = f2bf(a2.z); af[11] = f2bf(a2.w);
      af[12]= f2bf(a3.x); af[13]= f2bf(a3.y); af[14] = f2bf(a3.z); af[15] = f2bf(a3.w);
    }
    if (bValid0) {
      const float* bp = Bm + (long long)bCol0 * ldb + k0 + bKb;
      float4 b0 = *(const float4*)(bp);
      float4 b1 = *(const float4*)(bp + 4);
      float4 b2 = *(const float4*)(bp + 8);
      float4 b3 = *(const float4*)(bp + 12);
      bf0[0] = f2bf(b0.x); bf0[1] = f2bf(b0.y); bf0[2]  = f2bf(b0.z); bf0[3]  = f2bf(b0.w);
      bf0[4] = f2bf(b1.x); bf0[5] = f2bf(b1.y); bf0[6]  = f2bf(b1.z); bf0[7]  = f2bf(b1.w);
      bf0[8] = f2bf(b2.x); bf0[9] = f2bf(b2.y); bf0[10] = f2bf(b2.z); bf0[11] = f2bf(b2.w);
      bf0[12]= f2bf(b3.x); bf0[13]= f2bf(b3.y); bf0[14] = f2bf(b3.z); bf0[15] = f2bf(b3.w);
    } else {
      #pragma unroll
      for (int i = 0; i < 16; ++i) bf0[i] = zb;
    }
    if (bValid1) {
      const float* bp = Bm + (long long)bCol1 * ldb + k0 + bKb;
      float4 b0 = *(const float4*)(bp);
      float4 b1 = *(const float4*)(bp + 4);
      float4 b2 = *(const float4*)(bp + 8);
      float4 b3 = *(const float4*)(bp + 12);
      bf1[0] = f2bf(b0.x); bf1[1] = f2bf(b0.y); bf1[2]  = f2bf(b0.z); bf1[3]  = f2bf(b0.w);
      bf1[4] = f2bf(b1.x); bf1[5] = f2bf(b1.y); bf1[6]  = f2bf(b1.z); bf1[7]  = f2bf(b1.w);
      bf1[8] = f2bf(b2.x); bf1[9] = f2bf(b2.y); bf1[10] = f2bf(b2.z); bf1[11] = f2bf(b2.w);
      bf1[12]= f2bf(b3.x); bf1[13]= f2bf(b3.y); bf1[14] = f2bf(b3.z); bf1[15] = f2bf(b3.w);
    } else {
      #pragma unroll
      for (int i = 0; i < 16; ++i) bf1[i] = zb;
    }
    acc0 = __builtin_amdgcn_wmma_f32_16x16x32_bf16(false, af, false, bf0,
                                                   (short)0, acc0, false, false);
    acc1 = __builtin_amdgcn_wmma_f32_16x16x32_bf16(false, af, false, bf1,
                                                   (short)0, acc1, false, false);
  }

  const float bb0 = (bias != nullptr && bValid0) ? bias[bCol0] : 0.f;
  const float bb1 = (bias != nullptr && bValid1) ? bias[bCol1] : 0.f;
  #pragma unroll
  for (int r = 0; r < 8; ++r) {
    const int row = row0 + r + half * 8;
    if (bValid0) {
      float v = acc0[r] + bb0;
      if (relu) v = fmaxf(v, 0.f);
      C[(long long)row * ldc + bCol0] = v;
    }
    if (bValid1) {
      float v = acc1[r] + bb1;
      if (relu) v = fmaxf(v, 0.f);
      C[(long long)row * ldc + bCol1] = v;
    }
  }
}

// ---------------------------------------------------------------------------
// Transpose the V heads out of the packed qkv buffer:
//   vT[(b*NH+h)*HD + n][k] = qkv[b*QN + k][512 + h*HD + n]
// LDS-tiled 32x32 transpose (33-float row pitch kills bank conflicts); both
// the global read and write are fully coalesced.  This turns the attn @ V
// GEMM into the contiguous-operand WMMA path (no strided gathers).
// ---------------------------------------------------------------------------
__global__ __launch_bounds__(256)
void transpose_v_kernel(const float* __restrict__ qkv, float* __restrict__ vT)
{
  __shared__ float tile[32][33];
  const int z  = blockIdx.y;                 // b*NH + h
  const int b  = z >> 3;
  const int h  = z & 7;
  const int k0 = blockIdx.x * 32;
  const int tx = threadIdx.x & 31;
  const int ty = threadIdx.x >> 5;           // 8 rows per pass

  const float* srcp = qkv + (long long)b * CQN * (3 * CD) + 2 * CD + h * CHD;
  #pragma unroll
  for (int i = 0; i < 4; ++i) {
    const int k = k0 + ty + i * 8;
    tile[ty + i * 8][tx] = srcp[(long long)k * (3 * CD) + tx];
  }
  __syncthreads();
  float* dstp = vT + (long long)z * CHD * CQN;
  #pragma unroll
  for (int i = 0; i < 4; ++i) {
    const int n = ty + i * 8;
    dstp[(long long)n * CQN + k0 + tx] = tile[tx][n];
  }
}

// ---------------------------------------------------------------------------
// Exact top-k via full descending bitonic sort in LDS (one block per batch).
// Packed key: [mapped-score | ~index] so ties break toward the lowest index,
// matching jax.lax.top_k.  16384 x 8B = 128KB dynamic LDS (<320KB WGP LDS).
// ---------------------------------------------------------------------------
__global__ __launch_bounds__(1024)
void topk_sort_kernel(const float* __restrict__ score, int* __restrict__ sel_idx,
                      int n, int k)
{
  extern __shared__ unsigned long long smem[];
  score   += (long long)blockIdx.x * n;
  sel_idx += (long long)blockIdx.x * k;
  const int tid = threadIdx.x;

  for (int i = tid; i < n; i += 1024) {
    unsigned u = __builtin_bit_cast(unsigned, score[i]);
    unsigned m = (u & 0x80000000u) ? ~u : (u | 0x80000000u);   // order-preserving map
    smem[i] = ((unsigned long long)m << 32) | (unsigned)(~i);
  }
  __syncthreads();

  for (int kk = 2; kk <= n; kk <<= 1) {
    for (int j = kk >> 1; j > 0; j >>= 1) {
      for (int i = tid; i < n; i += 1024) {
        const int ixj = i ^ j;
        if (ixj > i) {
          const bool desc = ((i & kk) == 0);
          unsigned long long a = smem[i], b = smem[ixj];
          if (desc ? (a < b) : (a > b)) { smem[i] = b; smem[ixj] = a; }
        }
      }
      __syncthreads();
    }
  }
  for (int i = tid; i < k; i += 1024)
    sel_idx[i] = (int)(~(unsigned)(smem[i] & 0xffffffffu));
}

// ---------------------------------------------------------------------------
// Gather selected rows (vectorized float4); also builds q_src and qk = q+pos
// for the first CQN (= top-1024) rows.
// ---------------------------------------------------------------------------
__global__ __launch_bounds__(256)
void gather_kernel(const float4* __restrict__ src4, const float4* __restrict__ pos4,
                   const float* __restrict__ ref, const int* __restrict__ sel_idx,
                   float4* __restrict__ sel_src4, float4* __restrict__ sel_pos4,
                   float* __restrict__ sel_ref, float4* __restrict__ q_src4,
                   float4* __restrict__ qk4)
{
  const int t = blockIdx.x * 256 + threadIdx.x;      // CB*CFG*64
  if (t >= CB * CFG * (CD / 4)) return;
  const int d4  = t & 63;
  const int row = t >> 6;                            // b*CFG + i
  const int i   = row & (CFG - 1);
  const int b   = row >> 12;
  const int si  = sel_idx[row];
  const long long sb = ((long long)b * CHW + si) * (CD / 4) + d4;
  float4 s = src4[sb], p = pos4[sb];
  sel_src4[(long long)row * (CD / 4) + d4] = s;
  sel_pos4[(long long)row * (CD / 4) + d4] = p;
  if (i < CQN) {
    const long long q = ((long long)b * CQN + i) * (CD / 4) + d4;
    q_src4[q] = s;
    float4 a; a.x = s.x + p.x; a.y = s.y + p.y; a.z = s.z + p.z; a.w = s.w + p.w;
    qk4[q] = a;
  }
  if (d4 == 0) {
    const long long rb = ((long long)b * CHW + si) * 7;
    #pragma unroll
    for (int c = 0; c < 7; ++c) sel_ref[(long long)row * 7 + c] = ref[rb + c];
  }
}

// ---------------------------------------------------------------------------
// Row softmax (len up to a few K), scale applied pre-exp.  One block per row.
// ---------------------------------------------------------------------------
__global__ __launch_bounds__(256)
void softmax_rows_kernel(float* __restrict__ x, int rowLen, float scale)
{
  __shared__ float sm[256];
  const long long base = (long long)blockIdx.x * rowLen;
  const int tid = threadIdx.x;
  float m = -3.4e38f;
  for (int i = tid; i < rowLen; i += 256) m = fmaxf(m, x[base + i] * scale);
  sm[tid] = m; __syncthreads();
  for (int s = 128; s; s >>= 1) { if (tid < s) sm[tid] = fmaxf(sm[tid], sm[tid + s]); __syncthreads(); }
  m = sm[0]; __syncthreads();
  float sum = 0.f;
  for (int i = tid; i < rowLen; i += 256) {
    float e = expf(x[base + i] * scale - m);
    x[base + i] = e; sum += e;
  }
  sm[tid] = sum; __syncthreads();
  for (int s = 128; s; s >>= 1) { if (tid < s) sm[tid] += sm[tid + s]; __syncthreads(); }
  const float inv = 1.f / sm[0];
  for (int i = tid; i < rowLen; i += 256) x[base + i] *= inv;
}

// Per-head softmax over the 25 sampling weights (row layout [NH][NP]).
__global__ void softmax25_kernel(float* __restrict__ aw, int nRows)
{
  const int t = blockIdx.x * 256 + threadIdx.x;
  if (t >= nRows) return;
  const int h = t % CNH;
  const int r = t / CNH;
  float* p = aw + (long long)r * (CNH * CNP) + h * CNP;
  float m = -3.4e38f;
  #pragma unroll
  for (int i = 0; i < CNP; ++i) m = fmaxf(m, p[i]);
  float s = 0.f;
  #pragma unroll
  for (int i = 0; i < CNP; ++i) { float e = expf(p[i] - m); p[i] = e; s += e; }
  const float inv = 1.f / s;
  #pragma unroll
  for (int i = 0; i < CNP; ++i) p[i] *= inv;
}

// ---------------------------------------------------------------------------
// LayerNorm over D=256 with fused residual: out = LN(x + res)*g + b.
// outRow = (row / rpbIn) * rpbOut + (row % rpbIn)  (lets qn write into the
// first 1024 rows of each batch of sel_src).
// ---------------------------------------------------------------------------
__global__ __launch_bounds__(256)
void layernorm_kernel(const float* __restrict__ x, const float* __restrict__ res,
                      const float* __restrict__ g, const float* __restrict__ bta,
                      float* __restrict__ out, int rpbIn, int rpbOut)
{
  __shared__ float sm[256];
  const int row = blockIdx.x;
  const long long outRow = (long long)(row / rpbIn) * rpbOut + (row % rpbIn);
  const int tid = threadIdx.x;
  const float v = x[(long long)row * CD + tid] + res[(long long)row * CD + tid];
  sm[tid] = v; __syncthreads();
  for (int s = 128; s; s >>= 1) { if (tid < s) sm[tid] += sm[tid + s]; __syncthreads(); }
  const float mean = sm[0] * (1.f / CD); __syncthreads();
  const float d = v - mean;
  sm[tid] = d * d; __syncthreads();
  for (int s = 128; s; s >>= 1) { if (tid < s) sm[tid] += sm[tid + s]; __syncthreads(); }
  const float var = sm[0] * (1.f / CD);
  out[outRow * CD + tid] = d * rsqrtf(var + 1e-5f) * g[tid] + bta[tid];
}

// ---------------------------------------------------------------------------
// Box-attention sampling: one wave per (b, query, head); lane = channel d.
// Computes rotated kernel grid + offsets, 25-point bilinear gather from v,
// weighted by the softmaxed attention weights.
// ---------------------------------------------------------------------------
__device__ __forceinline__ float sample_corner(const float* __restrict__ vb,
                                               int xi, int yi, float wgt)
{
  const bool valid = (xi >= 0) & (xi < CW) & (yi >= 0) & (yi < CH);
  const int xc = min(max(xi, 0), CW - 1);
  const int yc = min(max(yi, 0), CH - 1);
  const float s = vb[(long long)(yc * CW + xc) * CD];
  return valid ? s * wgt : 0.f;
}

__global__ __launch_bounds__(256)
void box_sample_kernel(const float* __restrict__ obox, const float* __restrict__ dgrid,
                       const float* __restrict__ aw, const float* __restrict__ selref,
                       const float* __restrict__ v, const float* __restrict__ kidx,
                       float* __restrict__ outp)
{
  const int wave = threadIdx.x >> 5;
  const int lane = threadIdx.x & 31;
  const int z = blockIdx.x * 8 + wave;                 // (b*CFG + l)*CNH + h
  if (z >= CB * CFG * CNH) return;
  const int h = z % CNH;
  const int row = z / CNH;                             // b*CFG + l
  const int b = row / CFG;

  const float* rw = selref + (long long)row * 7;
  const float cx = rw[0], cy = rw[1], w0 = rw[3], h0 = rw[4], ang = rw[6];
  const float* ob = obox + (long long)row * (CNH * 4) + h * 4;
  const float bx = cx + ob[0] * 0.125f * w0;
  const float by = cy + ob[1] * 0.125f * h0;
  const float bw = w0 + ob[2] * 0.125f * w0;
  const float bh = h0 + ob[3] * 0.125f * h0;
  const float ca = cosf(ang), sa = sinf(ang);
  const float sw = fmaxf(bw, 0.f), sh = fmaxf(bh, 0.f);

  const float* dg  = dgrid + (long long)row * (CNH * CNP * 2) + h * (CNP * 2);
  const float* awp = aw    + (long long)row * (CNH * CNP)     + h * CNP;
  const float* vb  = v + (long long)b * CHW * CD + h * CHD + lane;
  const float invHS = 1.f / 188.f;

  float acc = 0.f;
  #pragma unroll
  for (int p = 0; p < CNP; ++p) {
    const float fx = kidx[p * 2 + 0] * sw;
    const float fy = kidx[p * 2 + 1] * sh;
    const float gx = bx + ca * fx - sa * fy + dg[p * 2 + 0] * invHS;
    const float gy = by + sa * fx + ca * fy + dg[p * 2 + 1] * invHS;
    const float xf = gx * (float)CW - 0.5f;
    const float yf = gy * (float)CH - 0.5f;
    const float x0 = floorf(xf), y0 = floorf(yf);
    const float lx = xf - x0, ly = yf - y0;
    const int x0i = (int)x0, y0i = (int)y0;
    float s = 0.f;
    s += sample_corner(vb, x0i,     y0i,     (1.f - lx) * (1.f - ly));
    s += sample_corner(vb, x0i + 1, y0i,     lx * (1.f - ly));
    s += sample_corner(vb, x0i,     y0i + 1, (1.f - lx) * ly);
    s += sample_corner(vb, x0i + 1, y0i + 1, lx * ly);
    acc += awp[p] * s;
  }
  outp[(long long)row * CD + h * CHD + lane] = acc;
}

// ---------------------------------------------------------------------------
// Elementwise helpers (float4-vectorized).
// ---------------------------------------------------------------------------
__global__ void add4_kernel(const float4* __restrict__ a, const float4* __restrict__ b,
                            float4* __restrict__ o, long long n4)
{
  const long long t = (long long)blockIdx.x * 256 + threadIdx.x;
  if (t >= n4) return;
  float4 x = a[t], y = b[t];
  float4 r; r.x = x.x + y.x; r.y = x.y + y.y; r.z = x.z + y.z; r.w = x.w + y.w;
  o[t] = r;
}

__global__ void copy4_kernel(const float4* __restrict__ a, float4* __restrict__ o, long long n4)
{
  const long long t = (long long)blockIdx.x * 256 + threadIdx.x;
  if (t < n4) o[t] = a[t];
}

__global__ void scatter4_kernel(const float4* __restrict__ sel, const int* __restrict__ sel_idx,
                                float4* __restrict__ o)
{
  const int t = blockIdx.x * 256 + threadIdx.x;      // CB*CFG*64
  if (t >= CB * CFG * (CD / 4)) return;
  const int d4  = t & 63;
  const int row = t >> 6;
  const int b   = row >> 12;
  const int si  = sel_idx[row];
  o[((long long)b * CHW + si) * (CD / 4) + d4] = sel[(long long)row * (CD / 4) + d4];
}

// ---------------------------------------------------------------------------
// Host side
// ---------------------------------------------------------------------------
static void gemm(hipStream_t s,
                 const float* A, int lda,
                 const float* Bm, int ldb,
                 const float* bias, float* C, int ldc,
                 int M, int N, int K, int relu,
                 int nz = 1, int nz2 = 1,
                 long long a1 = 0, long long a2 = 0,
                 long long b1 = 0, long long b2 = 0,
                 long long c1 = 0, long long c2 = 0)
{
  dim3 grid((unsigned)((M + 127) / 128), (unsigned)((N + 31) / 32), (unsigned)nz);
  wmma_gemm_kernel<<<grid, 256, 0, s>>>(A, lda, Bm, ldb, bias, C, ldc,
                                        M, N, K, relu, nz2, a1, a2, b1, b2, c1, c2);
}

extern "C" void kernel_launch(void* const* d_in, const int* in_sizes, int n_in,
                              void* d_out, int out_size, void* d_ws, size_t ws_size,
                              hipStream_t stream)
{
  (void)in_sizes; (void)n_in; (void)out_size; (void)ws_size;

  const float* src         = (const float*)d_in[0];
  const float* pos         = (const float*)d_in[1];
  const float* ref_windows = (const float*)d_in[4];
  const float* score_mask  = (const float*)d_in[5];
  const float* in_proj_w   = (const float*)d_in[6];
  const float* in_proj_b   = (const float*)d_in[7];
  const float* mha_out_w   = (const float*)d_in[8];
  const float* mha_out_b   = (const float*)d_in[9];
  const float* value_w     = (const float*)d_in[10];
  const float* value_b     = (const float*)d_in[11];
  const float* lattn_w     = (const float*)d_in[12];
  const float* lattn_b     = (const float*)d_in[13];
  const float* lbox_w      = (const float*)d_in[14];
  const float* lbox_b      = (const float*)d_in[15];
  const float* soff_w      = (const float*)d_in[16];
  const float* soff_b      = (const float*)d_in[17];
  const float* caow        = (const float*)d_in[18];
  const float* caob        = (const float*)d_in[19];
  const float* lin1w       = (const float*)d_in[20];
  const float* lin1b       = (const float*)d_in[21];
  const float* lin2w       = (const float*)d_in[22];
  const float* lin2b       = (const float*)d_in[23];
  const float* qng         = (const float*)d_in[24];
  const float* qnb         = (const float*)d_in[25];
  const float* n1g         = (const float*)d_in[26];
  const float* n1b         = (const float*)d_in[27];
  const float* n2g         = (const float*)d_in[28];
  const float* n2b         = (const float*)d_in[29];
  const float* kidx        = (const float*)d_in[30];
  float* out = (float*)d_out;

  // workspace arena (256B-aligned slabs)
  char* wp = (char*)d_ws;
  auto alloc = [&](size_t nbytes) {
    void* p = (void*)wp;
    wp += (nbytes + 255) & ~(size_t)255;
    return p;
  };
  int*   sel_idx = (int*)  alloc(sizeof(int)   * (size_t)CB * CFG);
  float* sel_src = (float*)alloc(sizeof(float) * (size_t)CB * CFG * CD);
  float* sel_pos = (float*)alloc(sizeof(float) * (size_t)CB * CFG * CD);
  float* sel_ref = (float*)alloc(sizeof(float) * (size_t)CB * CFG * 7);
  float* q_src   = (float*)alloc(sizeof(float) * (size_t)CB * CQN * CD);
  float* qk      = (float*)alloc(sizeof(float) * (size_t)CB * CQN * CD);
  float* qkv     = (float*)alloc(sizeof(float) * (size_t)CB * CQN * (3 * CD));
  float* vT      = (float*)alloc(sizeof(float) * (size_t)CB * CNH * CHD * CQN);
  float* scores  = (float*)alloc(sizeof(float) * (size_t)CB * CNH * CQN * CQN);
  float* mha_ctx = (float*)alloc(sizeof(float) * (size_t)CB * CQN * CD);
  float* q2      = (float*)alloc(sizeof(float) * (size_t)CB * CQN * CD);
  float* query   = (float*)alloc(sizeof(float) * (size_t)CB * CFG * CD);
  float* vbuf    = (float*)alloc(sizeof(float) * (size_t)CB * CHW * CD);
  float* awbuf   = (float*)alloc(sizeof(float) * (size_t)CB * CFG * CNH * CNP);
  float* obox    = (float*)alloc(sizeof(float) * (size_t)CB * CFG * CNH * 4);
  float* dgridb  = (float*)alloc(sizeof(float) * (size_t)CB * CFG * CNH * CNP * 2);
  float* boxout  = (float*)alloc(sizeof(float) * (size_t)CB * CFG * CD);
  float* cabuf   = (float*)alloc(sizeof(float) * (size_t)CB * CFG * CD);
  float* ffn1    = (float*)alloc(sizeof(float) * (size_t)CB * CFG * CDFF);
  float* ffn2    = (float*)alloc(sizeof(float) * (size_t)CB * CFG * CD);

  // 1) exact top-4096 (sorted desc) per batch — 128KB dynamic LDS bitonic sort
  topk_sort_kernel<<<CB, 1024, (size_t)CHW * 8, stream>>>(score_mask, sel_idx, CHW, CFG);

  // 2) gather sel_src / sel_pos / sel_ref; q_src & qk = first 1024 rows
  {
    const int n = CB * CFG * (CD / 4);
    gather_kernel<<<(n + 255) / 256, 256, 0, stream>>>(
        (const float4*)src, (const float4*)pos, ref_windows, sel_idx,
        (float4*)sel_src, (float4*)sel_pos, sel_ref, (float4*)q_src, (float4*)qk);
  }

  // 3) MHA: qkv projections (q,k from qk; v from q_src) into [B*QN, 768]
  gemm(stream, qk,    CD, in_proj_w,            CD, in_proj_b,        qkv,        3 * CD,
       CB * CQN, 2 * CD, CD, 0);
  gemm(stream, q_src, CD, in_proj_w + 2*CD*CD,  CD, in_proj_b + 2*CD, qkv + 2*CD, 3 * CD,
       CB * CQN, CD, CD, 0);

  // 3b) transpose V heads into [b,h][HD][QN] so attn@V uses contiguous loads
  transpose_v_kernel<<<dim3(CQN / 32, CB * CNH), 256, 0, stream>>>(qkv, vT);

  // 4) scores[b,h] = Q Kt   (batched over z = b*NH+h)
  gemm(stream, qkv, 3 * CD, qkv + CD, 3 * CD, nullptr, scores, CQN,
       CQN, CQN, CHD, 0,
       CB * CNH, CNH,
       (long long)CQN * 3 * CD, CHD,
       (long long)CQN * 3 * CD, CHD,
       (long long)CNH * CQN * CQN, (long long)CQN * CQN);

  // 5) softmax(scores / sqrt(HD))
  softmax_rows_kernel<<<CB * CNH * CQN, 256, 0, stream>>>(scores, CQN, 0.1767766953f);

  // 6) ctx[b,h] = attn V   (vT is [N=HD, K=QN] row-major -> contiguous path)
  gemm(stream, scores, CQN, vT, CQN, nullptr, mha_ctx, CD,
       CQN, CHD, CQN, 0,
       CB * CNH, CNH,
       (long long)CNH * CQN * CQN, (long long)CQN * CQN,
       (long long)CNH * CHD * CQN, (long long)CHD * CQN,
       (long long)CQN * CD, CHD);

  // 7) MHA output projection
  gemm(stream, mha_ctx, CD, mha_out_w, CD, mha_out_b, q2, CD, CB * CQN, CD, CD, 0);

  // 8) qn layernorm -> writes straight into the first 1024 rows of sel_src
  layernorm_kernel<<<CB * CQN, 256, 0, stream>>>(q_src, q2, qng, qnb, sel_src, CQN, CFG);

  // 9) box-attention query = sel_src + sel_pos
  {
    const long long n4 = (long long)CB * CFG * CD / 4;
    add4_kernel<<<(unsigned)((n4 + 255) / 256), 256, 0, stream>>>(
        (const float4*)sel_src, (const float4*)sel_pos, (float4*)query, n4);
  }

  // 10) value projection over the full src (largest GEMM: 32768x256x256)
  gemm(stream, src, CD, value_w, CD, value_b, vbuf, CD, CB * CHW, CD, CD, 0);

  // 11) attention-weight / box / sampling-offset projections
  gemm(stream, query, CD, lattn_w, CD, lattn_b, awbuf, CNH * CNP,
       CB * CFG, CNH * CNP, CD, 0);
  softmax25_kernel<<<(CB * CFG * CNH + 255) / 256, 256, 0, stream>>>(awbuf, CB * CFG * CNH);
  gemm(stream, query, CD, lbox_w, CD, lbox_b, obox, CNH * 4,
       CB * CFG, CNH * 4, CD, 0);
  gemm(stream, query, CD, soff_w, CD, soff_b, dgridb, CNH * CNP * 2,
       CB * CFG, CNH * CNP * 2, CD, 0);

  // 12) rotated-box bilinear sampling + weighted sum
  box_sample_kernel<<<(CB * CFG * CNH) / 8, 256, 0, stream>>>(
      obox, dgridb, awbuf, sel_ref, vbuf, kidx, boxout);

  // 13) box-attention output projection + LN1
  gemm(stream, boxout, CD, caow, CD, caob, cabuf, CD, CB * CFG, CD, CD, 0);
  layernorm_kernel<<<CB * CFG, 256, 0, stream>>>(sel_src, cabuf, n1g, n1b, sel_src, CFG, CFG);

  // 14) FFN (ReLU fused into first GEMM) + LN2
  gemm(stream, sel_src, CD, lin1w, CD, lin1b, ffn1, CDFF, CB * CFG, CDFF, CD, 1);
  gemm(stream, ffn1, CDFF, lin2w, CDFF, lin2b, ffn2, CD, CB * CFG, CD, CDFF, 0);
  layernorm_kernel<<<CB * CFG, 256, 0, stream>>>(sel_src, ffn2, n2g, n2b, sel_src, CFG, CFG);

  // 15) out = src; out[b, sel_idx] = sel_src
  {
    const long long n4 = (long long)CB * CHW * CD / 4;
    copy4_kernel<<<(unsigned)((n4 + 255) / 256), 256, 0, stream>>>(
        (const float4*)src, (float4*)out, n4);
    const int n = CB * CFG * (CD / 4);
    scatter4_kernel<<<(n + 255) / 256, 256, 0, stream>>>(
        (const float4*)sel_src, sel_idx, (float4*)out);
  }
}
